// EdgeConv_67508295958884
// MI455X (gfx1250) — compile-verified, matrix operand order checked by reading, not run on hardware
//
#include <hip/hip_runtime.h>

// ---------------------------------------------------------------------------
// EdgeConv: h_theta = feat @ W_theta ; h_phi = feat @ W_phi
//           out[i]  = max_k h_theta[src_ind[i,k]] + (h_phi[i] - h_theta[i])
// N=65536, IN=OUT=512, K=16.  GEMMs via v_wmma_f32_16x16x32_bf16 (wave32).
// ---------------------------------------------------------------------------

typedef __attribute__((ext_vector_type(16))) __bf16 v16bf;
typedef __attribute__((ext_vector_type(8)))  __bf16 v8bf;
typedef __attribute__((ext_vector_type(8)))  float  v8f;

#define NROWS 65536
#define FEAT  512
#define NEIGH 16

// fp32 -> bf16, round-to-nearest-even (inputs finite, no NaN handling needed)
__device__ __forceinline__ unsigned int f2bf(float f) {
    unsigned int u = __float_as_uint(f);
    return (u + 0x7FFFu + ((u >> 16) & 1u)) >> 16;
}

// --------------------------- kernel 0a: feat -> bf16 ------------------------
__global__ __launch_bounds__(256) void convert_feat_kernel(
    const float* __restrict__ feat, unsigned int* __restrict__ featb_u32) {
    size_t base = ((size_t)blockIdx.x * 256 + threadIdx.x) * 8;
    const float4 a = *(const float4*)(feat + base);
    const float4 b = *(const float4*)(feat + base + 4);
    uint4 r;
    r.x = f2bf(a.x) | (f2bf(a.y) << 16);
    r.y = f2bf(a.z) | (f2bf(a.w) << 16);
    r.z = f2bf(b.x) | (f2bf(b.y) << 16);
    r.w = f2bf(b.z) | (f2bf(b.w) << 16);
    *(uint4*)(featb_u32 + base / 2) = r;
}

// ------------------ kernel 0b: W -> bf16, transposed to [N][K] --------------
__global__ __launch_bounds__(256) void convert_w_kernel(
    const float* __restrict__ Wt, const float* __restrict__ Wp,
    unsigned short* __restrict__ WtT, unsigned short* __restrict__ WpT) {
    int idx = blockIdx.x * 256 + threadIdx.x;   // 0 .. 512*512-1
    int k = idx >> 9;                           // row in W (K dim)
    int n = idx & 511;                          // col in W (N dim)
    WtT[n * FEAT + k] = (unsigned short)f2bf(Wt[idx]);
    WpT[n * FEAT + k] = (unsigned short)f2bf(Wp[idx]);
}

// -------- helper: load a 16x32 bf16 A/B fragment (W pre-transposed) ---------
// Lane L holds row (L&15): bytes [0..15] = K 0..7 (lanes<16) / K 8..15 (>=16),
// bytes [16..31] = +16 K.  Matches CDNA5 16-bit A/B VGPR layout.
// base is wave-uniform; off is a 32-bit per-lane element offset -> SADDR form.
__device__ __forceinline__ v16bf load_frag(const __bf16* __restrict__ base, int off) {
    v8bf lo = *(const v8bf*)(base + off);
    v8bf hi = *(const v8bf*)(base + off + 16);
    return __builtin_shufflevector(lo, hi, 0, 1, 2, 3, 4, 5, 6, 7,
                                           8, 9, 10, 11, 12, 13, 14, 15);
}

// --------------------- kernel 1: fused dual GEMM (WMMA) ---------------------
// Block: 256 threads = 8 waves, tile M=128 x N=128, both weight matrices.
// Wave grid 2(M) x 4(N): per wave 64 rows x 32 cols x {theta, phi}
//   -> 4 A-frags reused across 4 B-frags: 16 b128 loads per 16 WMMAs.
__global__ __launch_bounds__(256) void gemm_dual_kernel(
    const __bf16* __restrict__ featb,
    const __bf16* __restrict__ WtT,
    const __bf16* __restrict__ WpT,
    float* __restrict__ h_theta,      // ws
    float* __restrict__ h_dst) {      // d_out (phi - theta)
    const int lane = threadIdx.x & 31;
    const int wave = threadIdx.x >> 5;
    const int wm = wave & 1;          // 0..1  (M direction, 64 rows each)
    const int wn = wave >> 1;         // 0..3  (N direction, 32 cols each)
    const int bn = blockIdx.x * 128;  // x walks N (4 blocks share A rows in L2)
    const int bm = blockIdx.y * 128;
    const int lrow = lane & 15;
    const int lsel = (lane >> 4) << 3;      // 0 or 8 (K element offset)

    v8f ct[4][2], cp[4][2];
#pragma unroll
    for (int mt = 0; mt < 4; ++mt)
#pragma unroll
        for (int nt = 0; nt < 2; ++nt)
#pragma unroll
            for (int e = 0; e < 8; ++e) { ct[mt][nt][e] = 0.f; cp[mt][nt][e] = 0.f; }

    int aoff[4];
#pragma unroll
    for (int mt = 0; mt < 4; ++mt)
        aoff[mt] = (bm + wm * 64 + mt * 16 + lrow) * FEAT + lsel;
    int boff[2];
#pragma unroll
    for (int nt = 0; nt < 2; ++nt)
        boff[nt] = (bn + wn * 32 + nt * 16 + lrow) * FEAT + lsel;

    for (int kk = 0; kk < FEAT; kk += 32) {
        v16bf a[4];
#pragma unroll
        for (int mt = 0; mt < 4; ++mt) a[mt] = load_frag(featb, aoff[mt] + kk);
#pragma unroll
        for (int nt = 0; nt < 2; ++nt) {
            v16bf bt = load_frag(WtT, boff[nt] + kk);
            v16bf bp = load_frag(WpT, boff[nt] + kk);
#pragma unroll
            for (int mt = 0; mt < 4; ++mt) {
                ct[mt][nt] = __builtin_amdgcn_wmma_f32_16x16x32_bf16(
                    false, a[mt], false, bt, (short)0, ct[mt][nt], false, false);
                cp[mt][nt] = __builtin_amdgcn_wmma_f32_16x16x32_bf16(
                    false, a[mt], false, bp, (short)0, cp[mt][nt], false, false);
            }
        }
    }

    // Epilogue: C/D layout — elem e: row = e + (lane>=16 ? 8 : 0), col = lane&15
    const int rofs = (lane >> 4) << 3;
#pragma unroll
    for (int mt = 0; mt < 4; ++mt) {
#pragma unroll
        for (int nt = 0; nt < 2; ++nt) {
            int col = bn + wn * 32 + nt * 16 + lrow;
#pragma unroll
            for (int e = 0; e < 8; ++e) {
                int row = bm + wm * 64 + mt * 16 + e + rofs;
                size_t o = (size_t)row * FEAT + col;
                float t = ct[mt][nt][e];
                h_theta[o] = t;
                h_dst[o] = cp[mt][nt][e] - t;
            }
        }
    }
}

// ------------------- kernel 2: gather + 16-way max + add --------------------
// 2 rows per block; 128 threads per row, 4 floats per thread.
// h_theta (128MB) is L2-resident on MI455X (192MB L2) -> gather hits L2.
__global__ __launch_bounds__(256) void gather_max_kernel(
    const long long* __restrict__ src_ind,
    const float* __restrict__ h_theta,
    float* __restrict__ out) {        // d_out: holds h_dst on entry
    const int rib = threadIdx.x >> 7;       // row in block: 0..1
    const int t = threadIdx.x & 127;
    const int row = blockIdx.x * 2 + rib;

    __shared__ int sidx[2][NEIGH];
    if (t < NEIGH) sidx[rib][t] = (int)src_ind[(size_t)row * NEIGH + t];
    __syncthreads();

    const int col = t * 4;
    float4 acc;
    acc.x = acc.y = acc.z = acc.w = -__builtin_inff();
#pragma unroll
    for (int k = 0; k < NEIGH; ++k) {
        const float4 v = *(const float4*)(h_theta + (size_t)sidx[rib][k] * FEAT + col);
        acc.x = fmaxf(acc.x, v.x);
        acc.y = fmaxf(acc.y, v.y);
        acc.z = fmaxf(acc.z, v.z);
        acc.w = fmaxf(acc.w, v.w);
    }
    float4 d = *(const float4*)(out + (size_t)row * FEAT + col);
    acc.x += d.x; acc.y += d.y; acc.z += d.z; acc.w += d.w;
    *(float4*)(out + (size_t)row * FEAT + col) = acc;
}

// ---------------------------------------------------------------------------
extern "C" void kernel_launch(void* const* d_in, const int* in_sizes, int n_in,
                              void* d_out, int out_size, void* d_ws, size_t ws_size,
                              hipStream_t stream) {
    // inputs: [0]=k (scalar, ==16), [1]=src_ind int64, [2]=feat f32,
    //         [3]=W_theta f32, [4]=W_phi f32
    const long long* src_ind = (const long long*)d_in[1];
    const float* feat = (const float*)d_in[2];
    const float* W_theta = (const float*)d_in[3];
    const float* W_phi = (const float*)d_in[4];
    float* out = (float*)d_out;

    char* ws = (char*)d_ws;
    // workspace layout (bytes)
    __bf16* featb = (__bf16*)(ws);                                  //  64 MB
    unsigned short* WtT_u = (unsigned short*)(ws + 67108864);       // 512 KB
    unsigned short* WpT_u = (unsigned short*)(ws + 67633152);       // 512 KB
    float* h_theta = (float*)(ws + 68157440);                       // 128 MB

    // 0a: feat -> bf16   (65536*512 / 8 per thread / 256 per block)
    convert_feat_kernel<<<16384, 256, 0, stream>>>(feat, (unsigned int*)featb);
    // 0b: weights -> bf16 transposed [N][K]
    convert_w_kernel<<<1024, 256, 0, stream>>>(W_theta, W_phi, WtT_u, WpT_u);

    // 1: fused dual GEMM -> h_theta (ws), h_dst (d_out)
    //    grid.x = N blocks (consecutive blocks reuse same A rows via L2)
    dim3 g1(FEAT / 128, NROWS / 128);
    gemm_dual_kernel<<<g1, 256, 0, stream>>>(featb, (const __bf16*)WtT_u,
                                             (const __bf16*)WpT_u, h_theta, out);

    // 2: gather-max + add h_dst, in place on d_out
    gather_max_kernel<<<NROWS / 2, 256, 0, stream>>>(src_ind, h_theta, out);
}
